// EdgePredictor_2448131359386
// MI455X (gfx1250) — compile-verified, matrix operand order checked by reading, not run on hardware
//
#include <hip/hip_runtime.h>

#define EPSF 1e-6f

typedef __attribute__((ext_vector_type(16))) __bf16 v16bf;
typedef __attribute__((ext_vector_type(8)))  __bf16 v8bf;
typedef __attribute__((ext_vector_type(8)))  float  v8f;

// Build a 16-element bf16 WMMA fragment from two contiguous 16B LDS chunks:
// elems 0..7  = p[0..7]   (K = kb .. kb+7)
// elems 8..15 = p[16..23] (K = kb+16 .. kb+23)
static __device__ __forceinline__ v16bf frag16(const __bf16* p) {
    v8bf lo = *(const v8bf*)p;
    v8bf hi = *(const v8bf*)(p + 16);
    v16bf r;
#pragma unroll
    for (int i = 0; i < 8; ++i) { r[i] = lo[i]; r[i + 8] = hi[i]; }
    return r;
}

// ---------------- zero scratch accumulators ----------------
__global__ void k_zero(float* p, long n) {
    long i = (long)blockIdx.x * 256 + threadIdx.x;
    if (i < n) p[i] = 0.0f;
}

// ---------------- per-edge: ef = ||pos[c]-pos[r]||, deg histogram, ef stats ----------------
__global__ void k_edge(const int* ei, const float* pos, float* ef, int* degcnt,
                       float* efacc, int E) {
    int t = threadIdx.x;
    int e = blockIdx.x * 256 + t;
    float v = 0.0f, v2 = 0.0f;
    if (e < E) {
        int r = ei[e] - 1;
        int c = ei[E + e] - 1;
        float dx = pos[c * 3 + 0] - pos[r * 3 + 0];
        float dy = pos[c * 3 + 1] - pos[r * 3 + 1];
        float dz = pos[c * 3 + 2] - pos[r * 3 + 2];
        float d = sqrtf(dx * dx + dy * dy + dz * dz);
        ef[e] = d;
        atomicAdd(&degcnt[c], 1);
        v = d; v2 = d * d;
    }
    __shared__ float s[256], s2[256];
    s[t] = v; s2[t] = v2;
    __syncthreads();
    for (int off = 128; off; off >>= 1) {
        if (t < off) { s[t] += s[t + off]; s2[t] += s2[t + off]; }
        __syncthreads();
    }
    if (t == 0) { atomicAdd(&efacc[0], s[0]); atomicAdd(&efacc[1], s2[0]); }
}

// ---------------- vocab histogram of x ----------------
__global__ void k_xhist(const int* x, int* xcnt, int N) {
    int n = blockIdx.x * 256 + threadIdx.x;
    if (n < N) atomicAdd(&xcnt[x[n]], 1);
}

// ---------------- node/ef prep: h0 table (BN over emb[x]) + rank-1 u,t for ef block ----------------
__global__ void k_nodeprep(const float* emb, const float* g4, const float* be4,
                           const int* xcnt, const float* efacc,
                           const float* g3, const float* be3, const float* fc1w,
                           const float* w1, float* h0tab, float* uvec, float* tvec,
                           int N, int E, int V) {
    int f = threadIdx.x;  // 0..127
    // column stats of emb[x] via vocab counts
    float s = 0.0f, s2 = 0.0f;
    for (int v = 0; v < V; ++v) {
        float xv = emb[v * 128 + f];
        float cnt = (float)xcnt[v];
        s += cnt * xv; s2 += cnt * xv * xv;
    }
    float m = s / (float)N;
    float var = s2 / (float)N - m * m;
    float rstd = rsqrtf(var + EPSF);
    float gg = g4[f], bb = be4[f];
    for (int v = 0; v < V; ++v)
        h0tab[v * 128 + f] = gg * (emb[v * 128 + f] - m) * rstd + bb;
    // ef-BN affine per feature (fc1_b cancels inside BN)
    __shared__ float sa[128], sb[128];
    float me = efacc[0] / (float)E;
    float ve = efacc[1] / (float)E - me * me;
    float w = fc1w[f];
    float al = g3[f] * w * rsqrtf(w * w * ve + EPSF);
    sa[f] = al;
    sb[f] = be3[f] - al * me;
    __syncthreads();
    // rank-1 collapse of the ef block of GEMM1: hw1 += ef_e*u + t
    float u = 0.0f, tt = 0.0f;
    for (int k = 0; k < 128; ++k) {
        float wv = w1[(256 + k) * 128 + f];
        u += sa[k] * wv;
        tt += sb[k] * wv;
    }
    uvec[f] = u; tvec[f] = tt;
}

// ---------------- dinv = 1/sqrt(1 + col-degree) ----------------
__global__ void k_dinv(const int* degcnt, float* dinv, int N) {
    int j = blockIdx.x * 256 + threadIdx.x;
    if (j < N) dinv[j] = rsqrtf(1.0f + (float)degcnt[j]);
}

// ---------------- P = h0tab @ W1[0:128], Q = h0tab @ W1[128:256]  (tiny) ----------------
__global__ void k_pq(const float* h0tab, const float* w1, float* P, float* Q, int V) {
    int gid = blockIdx.x * 256 + threadIdx.x;
    if (gid >= V * 128) return;
    int v = gid >> 7, j = gid & 127;
    float p = 0.0f, q = 0.0f;
    for (int k = 0; k < 128; ++k) {
        float h = h0tab[v * 128 + k];
        p += h * w1[k * 128 + j];
        q += h * w1[(128 + k) * 128 + j];
    }
    P[gid] = p; Q[gid] = q;
}

// ---------------- hw1[e] = P[x[row]] + Q[x[col]] + ef_e*u + t  (GEMM1 collapsed) ----------------
__global__ void k_hw1(const int* ei, const int* x, const float* ef,
                      const float* P, const float* Q, const float* uvec, const float* tvec,
                      float* hw1, int E) {
    int t = threadIdx.x;
    int lane = t & 31;
    int e = blockIdx.x * 8 + (t >> 5);
    if (e >= E) return;
    int r = ei[e] - 1, c = ei[E + e] - 1;
    int xr = x[r], xc = x[c];
    float efe = ef[e];
    float4 pv = ((const float4*)(P + xr * 128))[lane];
    float4 qv = ((const float4*)(Q + xc * 128))[lane];
    float4 uv = ((const float4*)uvec)[lane];
    float4 tv = ((const float4*)tvec)[lane];
    float4 o;
    o.x = pv.x + qv.x + efe * uv.x + tv.x;
    o.y = pv.y + qv.y + efe * uv.y + tv.y;
    o.z = pv.z + qv.z + efe * uv.z + tv.z;
    o.w = pv.w + qv.w + efe * uv.w + tv.w;
    ((float4*)(hw1 + (long)e * 128))[lane] = o;
}

// ---------------- scatter: S1[col] += dinv[row]*hw1[row]  (128 feat) ----------------
__global__ void k_s1(const int* ei, const float* dinv, const float* hw1, float* S1, int E) {
    int t = threadIdx.x;
    int lane = t & 31;
    int e = blockIdx.x * 8 + (t >> 5);
    if (e >= E) return;
    int r = ei[e] - 1, c = ei[E + e] - 1;
    float w = dinv[r];
    const float* src = hw1 + (long)r * 128;
    float* dst = S1 + (long)c * 128;
#pragma unroll
    for (int i = 0; i < 4; ++i) {
        int f = lane * 4 + i;
        atomicAdd(&dst[f], w * src[f]);
    }
}

// ---------------- z1 = relu(self + agg + b1) in-place, + per-feature stats (128) ----------------
__global__ void k_z1(float* hw1, const float* S1, const float* dinv, const float* b1,
                     float* fsum, float* fssq, int E, int N) {
    __shared__ float ssum[256], sssq[256];
    int t = threadIdx.x;
    int f = t & 127;
    int half = t >> 7;
    long base = (long)blockIdx.x * 128;
    float b = b1[f];
    float ls = 0.0f, lq = 0.0f;
    for (int r = half; r < 128; r += 2) {
        long j = base + r;
        if (j >= E) break;
        float* pr = hw1 + j * 128 + f;
        float v = *pr;
        float z;
        if (j < N) {
            float dj = dinv[j];
            z = dj * dj * v + dj * S1[j * 128 + f] + b;
        } else {
            z = v + b;
        }
        z = fmaxf(z, 0.0f);
        *pr = z;
        ls += z; lq += z * z;
    }
    ssum[t] = ls; sssq[t] = lq;
    __syncthreads();
    if (t < 128) {
        atomicAdd(&fsum[f], ssum[t] + ssum[t + 128]);
        atomicAdd(&fssq[f], sssq[t] + sssq[t + 128]);
    }
}

// ---------------- BN coefficients: a = g*rstd, c = be - a*mean ----------------
__global__ void k_bncoef(const float* fsum, const float* fssq, const float* g, const float* be,
                         float* a, float* c, int F, float invE) {
    int f = blockIdx.x * blockDim.x + threadIdx.x;
    if (f >= F) return;
    float m = fsum[f] * invE;
    float var = fssq[f] * invE - m * m;
    float r = rsqrtf(var + EPSF);
    float af = g[f] * r;
    a[f] = af;
    c[f] = be[f] - af * m;
}

// ---------------- GEMM2: hw2[E,64] = (a1*z1+c1)[E,128] @ w2[128,64]  (bf16 WMMA) ----------------
__global__ void __launch_bounds__(256) k_gemm2(const float* zin, const float* w2,
                                               const float* a1, const float* c1,
                                               float* hw2) {
    __shared__ __align__(16) __bf16 sA[128 * 136];
    __shared__ __align__(16) __bf16 sB[64 * 136];   // transposed: sB[n*136 + k] = w2[k][n]
    __shared__ float sa[128], sc[128];
    int t = threadIdx.x;
    long base = (long)blockIdx.x * 128;
    if (t < 128) { sa[t] = a1[t]; sc[t] = c1[t]; }
#pragma unroll
    for (int i = 0; i < 32; ++i) {                  // 128*64 B elements
        int idx = t + i * 256;
        int k = idx >> 6, n = idx & 63;
        sB[n * 136 + k] = (__bf16)w2[idx];
    }
    __syncthreads();
#pragma unroll 4
    for (int i = 0; i < 64; ++i) {                  // 128*128 A elements, BN fused
        int idx = t + i * 256;
        int r = idx >> 7, f = idx & 127;
        float xv = zin[(base + r) * 128 + f];
        sA[r * 136 + f] = (__bf16)(sa[f] * xv + sc[f]);
    }
    __syncthreads();
    int lane = t & 31, lr = lane & 15, lhi = lane >> 4, wv = t >> 5;
    int arow = wv * 16 + lr;
    v8f acc0 = {}, acc1 = {}, acc2 = {}, acc3 = {};
#pragma unroll
    for (int ks = 0; ks < 4; ++ks) {
        int kb = ks * 32 + lhi * 8;
        v16bf af = frag16(&sA[arow * 136 + kb]);
        v16bf b0 = frag16(&sB[(0 + lr) * 136 + kb]);
        acc0 = __builtin_amdgcn_wmma_f32_16x16x32_bf16(false, af, false, b0, (short)0, acc0, false, false);
        v16bf b1f = frag16(&sB[(16 + lr) * 136 + kb]);
        acc1 = __builtin_amdgcn_wmma_f32_16x16x32_bf16(false, af, false, b1f, (short)0, acc1, false, false);
        v16bf b2f = frag16(&sB[(32 + lr) * 136 + kb]);
        acc2 = __builtin_amdgcn_wmma_f32_16x16x32_bf16(false, af, false, b2f, (short)0, acc2, false, false);
        v16bf b3f = frag16(&sB[(48 + lr) * 136 + kb]);
        acc3 = __builtin_amdgcn_wmma_f32_16x16x32_bf16(false, af, false, b3f, (short)0, acc3, false, false);
    }
    long orow = base + wv * 16 + lhi * 8;
#pragma unroll
    for (int i = 0; i < 8; ++i) {
        hw2[(orow + i) * 64 + 0  + lr] = acc0[i];
        hw2[(orow + i) * 64 + 16 + lr] = acc1[i];
        hw2[(orow + i) * 64 + 32 + lr] = acc2[i];
        hw2[(orow + i) * 64 + 48 + lr] = acc3[i];
    }
}

// ---------------- scatter: S2[col] += dinv[row]*hw2[row]  (64 feat) ----------------
__global__ void k_s2(const int* ei, const float* dinv, const float* hw2, float* S2, int E) {
    int t = threadIdx.x;
    int lane = t & 15;
    int e = blockIdx.x * 16 + (t >> 4);
    if (e >= E) return;
    int r = ei[e] - 1, c = ei[E + e] - 1;
    float w = dinv[r];
    const float* src = hw2 + (long)r * 64;
    float* dst = S2 + (long)c * 64;
#pragma unroll
    for (int i = 0; i < 4; ++i) {
        int f = lane * 4 + i;
        atomicAdd(&dst[f], w * src[f]);
    }
}

// ---------------- z2 = relu(self + agg + b2) in-place, + per-feature stats (64) ----------------
__global__ void k_z2(float* hw2, const float* S2, const float* dinv, const float* b2,
                     float* fsum, float* fssq, int E, int N) {
    __shared__ float ssum[256], sssq[256];
    int t = threadIdx.x;
    int f = t & 63;
    int quart = t >> 6;
    long base = (long)blockIdx.x * 128;
    float b = b2[f];
    float ls = 0.0f, lq = 0.0f;
    for (int r = quart; r < 128; r += 4) {
        long j = base + r;
        if (j >= E) break;
        float* pr = hw2 + j * 64 + f;
        float v = *pr;
        float z;
        if (j < N) {
            float dj = dinv[j];
            z = dj * dj * v + dj * S2[j * 64 + f] + b;
        } else {
            z = v + b;
        }
        z = fmaxf(z, 0.0f);
        *pr = z;
        ls += z; lq += z * z;
    }
    ssum[t] = ls; sssq[t] = lq;
    __syncthreads();
    if (t < 64) {
        atomicAdd(&fsum[f], ssum[t] + ssum[t + 64] + ssum[t + 128] + ssum[t + 192]);
        atomicAdd(&fssq[f], sssq[t] + sssq[t + 64] + sssq[t + 128] + sssq[t + 192]);
    }
}

// ---------------- GEMM3: out[E,16] = (a2*z2+c2)[E,64] @ fc2_w.T[64,16] + fc2_b  (bf16 WMMA) ----------------
__global__ void __launch_bounds__(256) k_gemm3(const float* zin, const float* fc2w,
                                               const float* fc2b, const float* a2,
                                               const float* c2, float* out) {
    __shared__ __align__(16) __bf16 sA[128 * 72];
    __shared__ __align__(16) __bf16 sB[16 * 72];    // sB[o*72 + k] = fc2_w[o][k] (already B^T)
    __shared__ float sa[64], sc[64], sbias[16];
    int t = threadIdx.x;
    long base = (long)blockIdx.x * 128;
    if (t < 64) { sa[t] = a2[t]; sc[t] = c2[t]; }
    if (t < 16) sbias[t] = fc2b[t];
#pragma unroll
    for (int i = 0; i < 4; ++i) {                   // 16*64 B elements
        int idx = t + i * 256;
        int o = idx >> 6, k = idx & 63;
        sB[o * 72 + k] = (__bf16)fc2w[idx];
    }
    __syncthreads();
#pragma unroll 4
    for (int i = 0; i < 32; ++i) {                  // 128*64 A elements, BN fused
        int idx = t + i * 256;
        int r = idx >> 6, f = idx & 63;
        float xv = zin[(base + r) * 64 + f];
        sA[r * 72 + f] = (__bf16)(sa[f] * xv + sc[f]);
    }
    __syncthreads();
    int lane = t & 31, lr = lane & 15, lhi = lane >> 4, wv = t >> 5;
    int arow = wv * 16 + lr;
    v8f acc = {};
#pragma unroll
    for (int ks = 0; ks < 2; ++ks) {
        int kb = ks * 32 + lhi * 8;
        v16bf af = frag16(&sA[arow * 72 + kb]);
        v16bf bf = frag16(&sB[lr * 72 + kb]);
        acc = __builtin_amdgcn_wmma_f32_16x16x32_bf16(false, af, false, bf, (short)0, acc, false, false);
    }
    long orow = base + wv * 16 + lhi * 8;
    float bias = sbias[lr];
#pragma unroll
    for (int i = 0; i < 8; ++i)
        out[(orow + i) * 16 + lr] = acc[i] + bias;
}

extern "C" void kernel_launch(void* const* d_in, const int* in_sizes, int n_in,
                              void* d_out, int out_size, void* d_ws, size_t ws_size,
                              hipStream_t stream) {
    const int*   x    = (const int*)d_in[0];
    const int*   ei   = (const int*)d_in[1];
    const float* pos  = (const float*)d_in[2];
    const float* emb  = (const float*)d_in[3];
    const float* fc1w = (const float*)d_in[4];
    // d_in[5] = fc1_b: cancels inside BatchNorm, unused
    const float* fc2w = (const float*)d_in[6];
    const float* fc2b = (const float*)d_in[7];
    const float* w1   = (const float*)d_in[8];
    const float* b1   = (const float*)d_in[9];
    const float* w2   = (const float*)d_in[10];
    const float* b2   = (const float*)d_in[11];
    const float* g1   = (const float*)d_in[12];
    const float* be1  = (const float*)d_in[13];
    const float* g2   = (const float*)d_in[14];
    const float* be2  = (const float*)d_in[15];
    const float* g3   = (const float*)d_in[16];
    const float* be3  = (const float*)d_in[17];
    const float* g4   = (const float*)d_in[18];
    const float* be4  = (const float*)d_in[19];
    float* out = (float*)d_out;

    const int N = in_sizes[0];        // 20000 nodes
    const int E = in_sizes[1] / 2;    // 320000 edges
    const int V = in_sizes[3] / 128;  // 100 vocab

    float* W = (float*)d_ws;
    // zeroed-every-call region first
    long off = 0;
    float* S1    = W + off; off += (long)N * 128;
    float* S2    = W + off; off += (long)N * 64;
    int*   degc  = (int*)(W + off); off += N;
    int*   xcnt  = (int*)(W + off); off += 128;
    float* fsum1 = W + off; off += 128;
    float* fssq1 = W + off; off += 128;
    float* fsum2 = W + off; off += 64;
    float* fssq2 = W + off; off += 64;
    float* efacc = W + off; off += 8;
    const long ZERO_N = off;
    // persistent-within-call scratch
    float* hw1   = W + off; off += (long)E * 128;
    float* hw2   = W + off; off += (long)E * 64;
    float* ef    = W + off; off += E;
    float* dinv  = W + off; off += N;
    float* h0tab = W + off; off += (long)V * 128;
    float* P     = W + off; off += (long)V * 128;
    float* Q     = W + off; off += (long)V * 128;
    float* uvec  = W + off; off += 128;
    float* tvec  = W + off; off += 128;
    float* a1    = W + off; off += 128;
    float* c1    = W + off; off += 128;
    float* a2    = W + off; off += 64;
    float* c2    = W + off; off += 64;

    const float invE = 1.0f / (float)E;

    k_zero<<<(int)((ZERO_N + 255) / 256), 256, 0, stream>>>(W, ZERO_N);
    k_edge<<<(E + 255) / 256, 256, 0, stream>>>(ei, pos, ef, degc, efacc, E);
    k_xhist<<<(N + 255) / 256, 256, 0, stream>>>(x, xcnt, N);
    k_nodeprep<<<1, 128, 0, stream>>>(emb, g4, be4, xcnt, efacc, g3, be3, fc1w, w1,
                                      h0tab, uvec, tvec, N, E, V);
    k_dinv<<<(N + 255) / 256, 256, 0, stream>>>(degc, dinv, N);
    k_pq<<<(V * 128 + 255) / 256, 256, 0, stream>>>(h0tab, w1, P, Q, V);
    k_hw1<<<(E + 7) / 8, 256, 0, stream>>>(ei, x, ef, P, Q, uvec, tvec, hw1, E);
    k_s1<<<(E + 7) / 8, 256, 0, stream>>>(ei, dinv, hw1, S1, E);
    k_z1<<<(E + 127) / 128, 256, 0, stream>>>(hw1, S1, dinv, b1, fsum1, fssq1, E, N);
    k_bncoef<<<1, 128, 0, stream>>>(fsum1, fssq1, g1, be1, a1, c1, 128, invE);
    k_gemm2<<<E / 128, 256, 0, stream>>>(hw1, w2, a1, c1, hw2);
    k_s2<<<(E + 15) / 16, 256, 0, stream>>>(ei, dinv, hw2, S2, E);
    k_z2<<<(E + 127) / 128, 256, 0, stream>>>(hw2, S2, dinv, b2, fsum2, fssq2, E, N);
    k_bncoef<<<1, 64, 0, stream>>>(fsum2, fssq2, g2, be2, a2, c2, 64, invE);
    k_gemm3<<<E / 128, 256, 0, stream>>>(hw2, fc2w, fc2b, a2, c2, out);
}
